// MHA_48988396978746
// MI455X (gfx1250) — matrix operand order repaired and checked by reading, not compiled, not run on hardware
//
#include <hip/hip_runtime.h>

// ---------------------------------------------------------------------------
// MHA forward for MI455X (gfx1250, wave32, WMMA, async-LDS pipeline).
//   d_out = [ out (2*2048*1024 f32) | score (2*16*2048*2048 f32) ]
// Workspace: Qh/Kh (f16, (b*s,1024)), Vt (f16, (b,h,dh,s)), Attn (f16,
// (b,h,s,dh)) -- 8MB each; plus f16 copies of Wq/Wk/Wv/Wo (2MB each).
// ---------------------------------------------------------------------------

typedef __attribute__((ext_vector_type(16))) _Float16 v16h;
typedef __attribute__((ext_vector_type(8)))  _Float16 v8h;
typedef __attribute__((ext_vector_type(4)))  _Float16 v4h;
typedef __attribute__((ext_vector_type(8)))  float    v8f;
typedef __attribute__((ext_vector_type(4)))  float    v4f;

#if __has_builtin(__builtin_amdgcn_global_load_async_to_lds_b128)
#define ASYNC_LDS 1
#else
#define ASYNC_LDS 0
#endif

#if ASYNC_LDS
// exact builtin parameter types: int __attribute__((vector_size(16))) * in
// addrspace(1) (global) / addrspace(3) (LDS)
typedef int v4i_b128 __attribute__((vector_size(16)));
typedef __attribute__((address_space(1))) v4i_b128* gptr_b128;
typedef __attribute__((address_space(3))) v4i_b128* lptr_b128;

__device__ __forceinline__ void async_cp16(void* lds_dst, const void* gsrc) {
    // GLOBAL_LOAD_ASYNC_TO_LDS_B128: per-lane 16B global -> LDS, ASYNCcnt
    __builtin_amdgcn_global_load_async_to_lds_b128(
        (gptr_b128)gsrc, (lptr_b128)lds_dst, 0, 0);
}
#endif

template <int N>
__device__ __forceinline__ void wait_async() {
#if __has_builtin(__builtin_amdgcn_s_wait_asynccnt)
    __builtin_amdgcn_s_wait_asynccnt(N);
#elif ASYNC_LDS
    asm volatile("s_wait_asynccnt %0" ::"i"(N) : "memory");
#endif
}

__device__ __forceinline__ v8f wmma32(v16h a, v16h b, v8f c) {
    // D = A(16x32) * B(32x16) + C, f32 accumulate
    return __builtin_amdgcn_wmma_f32_16x16x32_f16(
        false, a, false, b, (short)0, c, false, false);
}

// Build a v16h fragment from two 16-byte (8-half) aligned chunks.
__device__ __forceinline__ v16h load_frag8(const _Float16* p0, const _Float16* p1) {
    union { v16h v; v8h h[2]; } u;
    u.h[0] = *(const v8h*)p0;
    u.h[1] = *(const v8h*)p1;
    return u.v;
}

// ---------------------------------------------------------------------------
// f32 -> f16 bulk convert (weights), vectorized x4.
// ---------------------------------------------------------------------------
__global__ __launch_bounds__(256) void cvt_f32_f16_k(
    const float* __restrict__ src, _Float16* __restrict__ dst, int n)
{
    const int i = (blockIdx.x * 256 + threadIdx.x) * 4;
    if (i < n) {
        v4f s = *(const v4f*)(src + i);
        v4h d;
        d[0] = (_Float16)s[0]; d[1] = (_Float16)s[1];
        d[2] = (_Float16)s[2]; d[3] = (_Float16)s[3];
        *(v4h*)(dst + i) = d;
    }
}

// ---------------------------------------------------------------------------
// GEMM: OUT[M,N] = X[M,K] @ Wh[N,K]^T + bias[N]   (M=4096, N=1024, K=1024)
// Wh is pre-converted f16. XT = float (convert while staging) or f16 (async).
// VT = true: scatter result as Vt(b,h,dh,s) via an LDS transpose (V proj).
// Block: 256 threads (8 waves), tile M=64 x N=128, K-step 32,
// double-buffered async global->LDS staging when the toolchain supports it.
// ---------------------------------------------------------------------------
template <typename XT, typename OT, bool VT>
__global__ __launch_bounds__(256) void gemm_wmma_k(
    const XT* __restrict__ X, const _Float16* __restrict__ Wh,
    const float* __restrict__ bias, OT* __restrict__ OUT,
    int M, int N, int K)
{
    // stride 40 halves = 80B: 16B-aligned vector rows + bank rotation
    __shared__ _Float16 Xs[2][64 * 40];
    __shared__ _Float16 Ws[2][128 * 40];

    const int tid  = threadIdx.x;
    const int lane = tid & 31;
    const int wid  = tid >> 5;
    const int hi   = lane >> 4;       // 0: lanes 0-15, 1: lanes 16-31
    const int l15  = lane & 15;
    const int wm   = wid >> 2;        // 0..1 -> M offset wm*32
    const int wn   = wid & 3;         // 0..3 -> N offset wn*32
    const int Mb   = blockIdx.y * 64;
    const int Nb   = blockIdx.x * 128;

    v8f c[2][2] = {};

    const int xrow = tid >> 2, xcol = (tid & 3) * 8;   // X: 64x32, 16B/thread
    // W: 128x32 = 512 x 16B chunks, 2 per thread
    const int wr0 = (tid * 2) >> 2,      wc0 = ((tid * 2) & 3) * 8;
    const int wr1 = (tid * 2 + 1) >> 2,  wc1 = ((tid * 2 + 1) & 3) * 8;

#if ASYNC_LDS
    auto issue_stage = [&](int k0, int bufi) {
        async_cp16(&Ws[bufi][wr0 * 40 + wc0], Wh + (size_t)(Nb + wr0) * K + k0 + wc0);
        async_cp16(&Ws[bufi][wr1 * 40 + wc1], Wh + (size_t)(Nb + wr1) * K + k0 + wc1);
        if constexpr (sizeof(XT) == 2) {
            async_cp16(&Xs[bufi][xrow * 40 + xcol],
                       (const _Float16*)X + (size_t)(Mb + xrow) * K + k0 + xcol);
        } else {
            const float* xp = (const float*)X + (size_t)(Mb + xrow) * K + k0 + xcol;
#pragma unroll
            for (int j = 0; j < 8; ++j)
                Xs[bufi][xrow * 40 + xcol + j] = (_Float16)xp[j];
        }
    };
    constexpr int OPS = (sizeof(XT) == 2) ? 3 : 2;   // async ops/stage/wave
    issue_stage(0, 0);
#endif

    for (int k0 = 0, s = 0; k0 < K; k0 += 32, ++s) {
        const int cur = s & 1;
#if ASYNC_LDS
        if (k0 + 32 < K) { issue_stage(k0 + 32, cur ^ 1); wait_async<OPS>(); }
        else             { wait_async<0>(); }
#else
        // synchronous staging fallback
        if constexpr (sizeof(XT) == 2) {
            const _Float16* xp = (const _Float16*)X + (size_t)(Mb + xrow) * K + k0 + xcol;
            *(v8h*)&Xs[cur][xrow * 40 + xcol] = *(const v8h*)xp;
        } else {
            const float* xp = (const float*)X + (size_t)(Mb + xrow) * K + k0 + xcol;
#pragma unroll
            for (int j = 0; j < 8; ++j)
                Xs[cur][xrow * 40 + xcol + j] = (_Float16)xp[j];
        }
        *(v8h*)&Ws[cur][wr0 * 40 + wc0] =
            *(const v8h*)(Wh + (size_t)(Nb + wr0) * K + k0 + wc0);
        *(v8h*)&Ws[cur][wr1 * 40 + wc1] =
            *(const v8h*)(Wh + (size_t)(Nb + wr1) * K + k0 + wc1);
#endif
        __syncthreads();

        const _Float16* xsb = Xs[cur];
        const _Float16* wsb = Ws[cur];
        v16h a[2], bb[2];
#pragma unroll
        for (int mt = 0; mt < 2; ++mt) {
            const _Float16* xr = &xsb[(wm * 32 + mt * 16 + l15) * 40];
            // A layout: half i -> K = hi*16 + i ; M = l15
            a[mt] = load_frag8(xr + hi * 16, xr + hi * 16 + 8);
        }
#pragma unroll
        for (int nt = 0; nt < 2; ++nt) {
            const _Float16* wr = &wsb[(wn * 32 + nt * 16 + l15) * 40 + hi * 16];
            // B layout: half i -> K = hi*16 + i ; N = l15
            bb[nt] = load_frag8(wr, wr + 8);
        }
#pragma unroll
        for (int mt = 0; mt < 2; ++mt)
#pragma unroll
            for (int nt = 0; nt < 2; ++nt)
                c[mt][nt] = wmma32(a[mt], bb[nt], c[mt][nt]);
        __syncthreads();
    }

    if constexpr (!VT) {
        // ---- plain epilogue: bias + store (C layout: M = e+8*hi, N = l15) ----
#pragma unroll
        for (int mt = 0; mt < 2; ++mt)
#pragma unroll
            for (int nt = 0; nt < 2; ++nt)
#pragma unroll
                for (int e = 0; e < 8; ++e) {
                    int r   = Mb + wm * 32 + mt * 16 + e + 8 * hi;
                    int col = Nb + wn * 32 + nt * 16 + l15;
                    OUT[(size_t)r * N + col] = (OT)(c[mt][nt][e] + bias[col]);
                }
    } else {
        // ---- V epilogue: transpose 64x128 tile through LDS, emit Vt(b,h,dh,s)
        __shared__ _Float16 Ts[64 * 136];   // pad 136 to rotate banks
#pragma unroll
        for (int mt = 0; mt < 2; ++mt)
#pragma unroll
            for (int nt = 0; nt < 2; ++nt)
#pragma unroll
                for (int e = 0; e < 8; ++e) {
                    int ml = wm * 32 + mt * 16 + e + 8 * hi;     // local s
                    int nl = wn * 32 + nt * 16 + l15;            // local col
                    Ts[ml * 136 + nl] = (_Float16)(c[mt][nt][e] + bias[Nb + nl]);
                }
        __syncthreads();
        const int c0 = tid >> 1, sc = (tid & 1) * 32;     // col 0..127, s-chunk
        const int colg = Nb + c0;
        const int hh = colg >> 6, dh = colg & 63;
        const int bb2 = Mb >> 11;                         // batch
        const int sbase = (Mb & 2047) + sc;
        _Float16* dst = (_Float16*)OUT +
            (((size_t)(bb2 * 16 + hh) * 64 + dh) * 2048 + sbase);
        _Float16 tmp[32];
#pragma unroll
        for (int j = 0; j < 32; ++j) tmp[j] = Ts[(sc + j) * 136 + c0];
#pragma unroll
        for (int j = 0; j < 32; j += 8)
            *(v8h*)(dst + j) = *(const v8h*)&tmp[j];
    }
}

// ---------------------------------------------------------------------------
// Fused attention: per block = one (b, h, 16-row q tile).
//   S = scale * Qh Kh^T -> mask -> softmax -> score (f32, NT-written once)
//   Attn = P @ Vt       -> f16 workspace in (b,h,s,dh) layout
// 8 waves: each wave owns a 256-wide kv stripe for the matmuls and 2 full
// rows for the softmax (wave32 shfl reductions).
// Dynamic LDS: Ss (16x2048 f32) | Ps (16x2048 f16) | acc (16x64 f32)
// ---------------------------------------------------------------------------
__global__ __launch_bounds__(256) void attn_fused_k(
    const _Float16* __restrict__ Qh, const _Float16* __restrict__ Kh,
    const _Float16* __restrict__ Vt, const int* __restrict__ mask,
    float* __restrict__ score_out, _Float16* __restrict__ attn_out)
{
    constexpr int S = 2048, H = 16, DH = 64, DM = 1024;
    const int tid  = threadIdx.x;
    const int lane = tid & 31;
    const int wid  = tid >> 5;
    const int hi   = lane >> 4;
    const int l15  = lane & 15;

    const int qt = blockIdx.x & 127;
    const int h  = (blockIdx.x >> 7) & 15;
    const int b  = blockIdx.x >> 11;
    const int qbase = qt * 16;

    extern __shared__ char smem[];
    float*    Ss  = (float*)smem;                                   // 131072 B
    _Float16* Ps  = (_Float16*)(smem + 16 * 2048 * 4);              //  65536 B
    float*    acc = (float*)(smem + 16 * 2048 * 4 + 16 * 2048 * 2); //   4096 B

    for (int i = tid; i < 16 * 64; i += 256) acc[i] = 0.0f;

    // ---- Q A-fragments (dh split into two k=32 chunks), reused by all tiles
    const _Float16* qrow = Qh + ((size_t)(b * S + qbase + l15)) * DM + h * DH;
    const v16h aq0 = load_frag8(qrow + hi * 16,      qrow + hi * 16 + 8);
    const v16h aq1 = load_frag8(qrow + 32 + hi * 16, qrow + 40 + hi * 16);

    const float scale = 0.125f;  // 1/sqrt(64)

    // ================= Phase 1: masked scaled scores -> LDS ==================
    for (int nt = 0; nt < 16; ++nt) {
        const int nb = wid * 256 + nt * 16;
        const _Float16* krow = Kh + ((size_t)(b * S + nb + l15)) * DM + h * DH + hi * 16;
        v16h bk0 = load_frag8(krow,      krow + 8);    // dh  0..31
        v16h bk1 = load_frag8(krow + 32, krow + 40);   // dh 32..63
        v8f c = {};
        c = wmma32(aq0, bk0, c);
        c = wmma32(aq1, bk1, c);
#pragma unroll
        for (int e = 0; e < 8; ++e) {
            const int qr = e + 8 * hi;
            const int kv = nb + l15;
            float s = c[e] * scale;
            const int mv = __builtin_nontemporal_load(
                &mask[((size_t)(b * S + qbase + qr)) * S + kv]);
            if (mv == 0) s = -1e9f;
            Ss[qr * 2048 + kv] = s;
        }
    }
    __syncthreads();

    // ====== Softmax: each wave owns 2 rows, wave32 shfl reductions ===========
    const size_t so_base = ((size_t)((b * H + h) * S) + qbase) * S;
#pragma unroll
    for (int rr = 0; rr < 2; ++rr) {
        const int row = wid * 2 + rr;
        float lmax = -3.0e38f;
        for (int i = 0; i < 16; ++i) {
            v4f s4 = *(const v4f*)&Ss[row * 2048 + lane * 4 + 128 * i];
            lmax = fmaxf(fmaxf(fmaxf(lmax, s4[0]), s4[1]), fmaxf(s4[2], s4[3]));
        }
#pragma unroll
        for (int off = 16; off >= 1; off >>= 1)
            lmax = fmaxf(lmax, __shfl_xor(lmax, off, 32));

        float lsum = 0.0f;
        for (int i = 0; i < 16; ++i) {
            const int idx = row * 2048 + lane * 4 + 128 * i;
            v4f s4 = *(const v4f*)&Ss[idx];
            v4f e4;
            e4[0] = __expf(s4[0] - lmax);
            e4[1] = __expf(s4[1] - lmax);
            e4[2] = __expf(s4[2] - lmax);
            e4[3] = __expf(s4[3] - lmax);
            *(v4f*)&Ss[idx] = e4;
            lsum += e4[0] + e4[1] + e4[2] + e4[3];
        }
#pragma unroll
        for (int off = 16; off >= 1; off >>= 1)
            lsum += __shfl_xor(lsum, off, 32);
        const float inv = 1.0f / lsum;

        for (int i = 0; i < 16; ++i) {
            const int kv = lane * 4 + 128 * i;
            v4f e4 = *(const v4f*)&Ss[row * 2048 + kv];
            v4f p4 = e4 * inv;
            // score is write-once: non-temporal, keep L2 for Kh/Vt reuse
            __builtin_nontemporal_store(
                p4, (v4f*)&score_out[so_base + (size_t)row * S + kv]);
            v4h ph;
            ph[0] = (_Float16)p4[0]; ph[1] = (_Float16)p4[1];
            ph[2] = (_Float16)p4[2]; ph[3] = (_Float16)p4[3];
            *(v4h*)&Ps[row * 2048 + kv] = ph;
        }
    }
    __syncthreads();

    // ================= Phase 2: Attn = P @ Vt ================================
    v8f co[4] = {};
    for (int kc = 0; kc < 8; ++kc) {
        const int kvb = wid * 256 + kc * 32;
        const _Float16* prow = Ps + l15 * 2048 + kvb;
        v16h ap = load_frag8(prow + hi * 16, prow + hi * 16 + 8);
#pragma unroll
        for (int nt = 0; nt < 4; ++nt) {
            // Vt(b,h,dh,s): B frag is two contiguous 16B loads per lane
            const _Float16* vtp = Vt +
                (((size_t)(b * H + h) * DH + nt * 16 + l15) * S) + kvb + 16 * hi;
            __builtin_prefetch(vtp + 256, 0, 0);   // next kv stripe chunk
            v16h bvv = load_frag8(vtp, vtp + 8);
            co[nt] = wmma32(ap, bvv, co[nt]);
        }
    }
    // cross-wave reduction of partial 16x64 tiles via LDS float atomics
#pragma unroll
    for (int nt = 0; nt < 4; ++nt)
#pragma unroll
        for (int e = 0; e < 8; ++e)
            atomicAdd(&acc[(e + 8 * hi) * 64 + nt * 16 + l15], co[nt][e]);
    __syncthreads();

    for (int i = tid; i < 16 * 64; i += 256) {
        const int r = i >> 6, ccol = i & 63;
        attn_out[((size_t)((b * H + h) * S) + qbase + r) * DH + ccol] =
            (_Float16)acc[i];
    }
}

// ---------------------------------------------------------------------------
extern "C" void kernel_launch(void* const* d_in, const int* in_sizes, int n_in,
                              void* d_out, int out_size, void* d_ws, size_t ws_size,
                              hipStream_t stream)
{
    (void)in_sizes; (void)n_in; (void)out_size; (void)ws_size;

    const float* q    = (const float*)d_in[0];
    const float* k    = (const float*)d_in[1];
    const float* v    = (const float*)d_in[2];
    const int*   mask = (const int*)  d_in[3];
    const float* Wq   = (const float*)d_in[4];
    const float* bq   = (const float*)d_in[5];
    const float* Wk   = (const float*)d_in[6];
    const float* bk   = (const float*)d_in[7];
    const float* Wv   = (const float*)d_in[8];
    const float* bv   = (const float*)d_in[9];
    const float* Wo   = (const float*)d_in[10];
    const float* bo   = (const float*)d_in[11];

    float* out   = (float*)d_out;                       // (2,2048,1024)
    float* score = out + (size_t)2 * 2048 * 1024;       // (2,16,2048,2048)

    _Float16* Qh  = (_Float16*)d_ws;                    // (b*s, 1024) f16
    _Float16* Kh  = Qh  + (size_t)4096 * 1024;
    _Float16* Vt  = Kh  + (size_t)4096 * 1024;          // (b,h,dh,s) f16
    _Float16* At  = Vt  + (size_t)4096 * 1024;          // (b,h,s,dh) f16
    _Float16* Wqh = At  + (size_t)4096 * 1024;          // f16 weights
    _Float16* Wkh = Wqh + (size_t)1024 * 1024;
    _Float16* Wvh = Wkh + (size_t)1024 * 1024;
    _Float16* Woh = Wvh + (size_t)1024 * 1024;

    // one-time f32 -> f16 weight conversion (2MB each)
    const int wn = 1024 * 1024;
    cvt_f32_f16_k<<<wn / (256 * 4), 256, 0, stream>>>(Wq, Wqh, wn);
    cvt_f32_f16_k<<<wn / (256 * 4), 256, 0, stream>>>(Wk, Wkh, wn);
    cvt_f32_f16_k<<<wn / (256 * 4), 256, 0, stream>>>(Wv, Wvh, wn);
    cvt_f32_f16_k<<<wn / (256 * 4), 256, 0, stream>>>(Wo, Woh, wn);

    const dim3 gg(1024 / 128, 4096 / 64);               // N tiles x M tiles

    gemm_wmma_k<float, _Float16, false><<<gg, 256, 0, stream>>>(q, Wqh, bq, Qh, 4096, 1024, 1024);
    gemm_wmma_k<float, _Float16, false><<<gg, 256, 0, stream>>>(k, Wkh, bk, Kh, 4096, 1024, 1024);
    gemm_wmma_k<float, _Float16, true ><<<gg, 256, 0, stream>>>(v, Wvh, bv, Vt, 4096, 1024, 1024);

    const size_t smem = 16 * 2048 * 4 + 16 * 2048 * 2 + 16 * 64 * 4;
    attn_fused_k<<<2 * 16 * 128, 256, smem, stream>>>(Qh, Kh, Vt, mask, score, At);

    // raw (b,h,s,dh) -> (b*2048, 1024) reinterpretation matches the
    // reference's non-transposed .reshape, so this is a plain GEMM.
    gemm_wmma_k<_Float16, float, false><<<gg, 256, 0, stream>>>(At, Woh, bo, out, 4096, 1024, 1024);
}
